// MaskEmbdMultiMPN_NNConv_39728447488423
// MI455X (gfx1250) — matrix operand order repaired and checked by reading, not compiled, not run on hardware
//
#include <hip/hip_runtime.h>

// ---------------------------------------------------------------------------
// MaskEmbdMultiMPN (NNConv GNN) for MI455X / gfx1250, wave32 + WMMA.
//
// Per 16-edge tile:  msg = Z @ B  with
//   Z[e,(k,i)] = h[e,k] * x[src_e, i]    (h recomputed from 2 edge attrs)
//   B[(k,i),o] = w2[k, i*FO+o] (k<64) ;  b2[i*FO+o] (k==64, h := 1)
// as a chain of v_wmma_f32_16x16x32_f16; the C tile IS msg[16,FO] and is
// scatter-added (segment_sum) with global_atomic_add_f32.  Z is built at
// f16-pair granularity (v_pk_mul_f16) to keep the VALU:WMMA ratio low.
// ---------------------------------------------------------------------------

#define N_NODES 50000
#define E_EDGES 400000   // undirected (2 * E_DIR)
#define NF_DIM  4
#define HID_DIM 16
#define OUT_DIM 4
#define EHID    64

typedef __attribute__((ext_vector_type(16))) _Float16 v16h;
typedef __attribute__((ext_vector_type(2)))  _Float16 v2h;
typedef __attribute__((ext_vector_type(8)))  float    v8f;

__device__ __forceinline__ float reluf(float v) { return fmaxf(v, 0.0f); }

// h_k = relu(ea0*w1[0,k] + ea1*w1[1,k] + b1[k]); k==EHID is the synthetic
// bias row (h=1) that folds b2 into the GEMM; k>EHID is K padding.
__device__ __forceinline__ float hval(int k, float ea0, float ea1,
                                      const float* __restrict__ w1,
                                      const float* __restrict__ b1)
{
    if (k < EHID)
        return fmaxf(fmaf(ea1, w1[EHID + k], fmaf(ea0, w1[k], b1[k])), 0.0f);
    if (k == EHID)
        return 1.0f;
    return 0.0f;
}

// ---------------------------------------------------------------------------
// x = x + relu(pred_mask @ mw1 + mb1) @ mw2 + mb2 ; pad rows to stride 16
// ---------------------------------------------------------------------------
__global__ void __launch_bounds__(256)
mask_embd_kernel(const float* __restrict__ x, const float* __restrict__ pm,
                 const float* __restrict__ mw1, const float* __restrict__ mb1,
                 const float* __restrict__ mw2, const float* __restrict__ mb2,
                 float* __restrict__ xout, int n)
{
    int i = blockIdx.x * blockDim.x + threadIdx.x;
    if (i >= n) return;
    float p0 = pm[i * 4 + 0], p1 = pm[i * 4 + 1];
    float p2 = pm[i * 4 + 2], p3 = pm[i * 4 + 3];
    float acc[4];
#pragma unroll
    for (int o = 0; o < 4; ++o) acc[o] = mb2[o];
#pragma unroll
    for (int j = 0; j < 16; ++j) {
        float h = mb1[j];
        h = fmaf(p0, mw1[0 * 16 + j], h);
        h = fmaf(p1, mw1[1 * 16 + j], h);
        h = fmaf(p2, mw1[2 * 16 + j], h);
        h = fmaf(p3, mw1[3 * 16 + j], h);
        h = reluf(h);
#pragma unroll
        for (int o = 0; o < 4; ++o) acc[o] = fmaf(h, mw2[j * 4 + o], acc[o]);
    }
#pragma unroll
    for (int o = 0; o < 4; ++o) xout[i * 16 + o] = x[i * 4 + o] + acc[o];
#pragma unroll
    for (int o = 4; o < 16; ++o) xout[i * 16 + o] = 0.0f;
}

__global__ void __launch_bounds__(256)
fill_zero_kernel(float* __restrict__ p, int n)
{
    int i = blockIdx.x * blockDim.x + threadIdx.x;
    if (i < n) p[i] = 0.0f;
}

// ---------------------------------------------------------------------------
// Fused edge kernel: one wave handles 16 edges; K = (EHID+1)*FI padded to 32.
// ---------------------------------------------------------------------------
template <int FI, int FO>
__global__ void __launch_bounds__(256)
edge_msg_kernel(const float* __restrict__ x,    // [N,16] (first FI cols valid)
                const int*   __restrict__ src,
                const int*   __restrict__ dst,
                const float* __restrict__ ea,   // [E,2]
                const float* __restrict__ w1,   // [2,64]
                const float* __restrict__ b1,   // [64]
                const float* __restrict__ w2,   // [64, FI*FO]
                const float* __restrict__ b2,   // [FI*FO]
                float*       __restrict__ agg,  // [N,16]
                int nEdges)
{
    constexpr int SH = (FI == 16) ? 4 : 2;       // log2(FI)
    constexpr int KV = (EHID + 1) * FI;          // valid K (incl. bias row)
    constexpr int KP = (KV + 31) & ~31;          // padded K
    constexpr int NSTEP = KP / 32;

    // B^T staged in LDS, f16, [16 cols][KP]; rows 32B-aligned.
    __shared__ alignas(32) _Float16 w2zT[16 * KP];
    for (int idx = threadIdx.x; idx < 16 * KP; idx += blockDim.x) {
        int o = idx / KP;
        int K = idx - o * KP;
        int k = K >> SH, i = K & (FI - 1);
        float v = 0.0f;
        if (o < FO && K < KV)
            v = (k < EHID) ? w2[(k * FI + i) * FO + o] : b2[i * FO + o];
        w2zT[idx] = (_Float16)v;
    }
    __syncthreads();

    const int lane  = threadIdx.x & 31;
    const int lhalf = lane >> 4;                 // 0 / 1
    const int erow  = lane & 15;                 // A-row (edge) & B/C column
    const int ihofs = lhalf << 3;                // K offset within a 16-run
    const int iofs  = (FI == 16) ? ihofs : 0;    // x-feature offset this lane

    const int wavesPerBlock = blockDim.x >> 5;
    const int wid  = blockIdx.x * wavesPerBlock + (threadIdx.x >> 5);
    const int nwav = gridDim.x * wavesPerBlock;
    const int nTiles = (nEdges + 15) >> 4;

    const _Float16* brow = &w2zT[erow * KP + (lhalf << 3)];

    for (int t = wid; t < nTiles; t += nwav) {
        int e = t * 16 + erow;
        bool ev = (e < nEdges);
        int ec = ev ? e : 0;
        int srow = src[ec];
        float ea0 = ea[ec * 2 + 0];
        float ea1 = ea[ec * 2 + 1];

        // x[src] slice for this lane as 4 packed f16 pairs
        const float* xrow = x + (size_t)srow * 16 + iofs;
        float4 q0 = *(const float4*)(xrow);
        float4 q1 = (FI == 16) ? *(const float4*)(xrow + 4)
                               : make_float4(0.f, 0.f, 0.f, 0.f);
        if (!ev) { q0 = make_float4(0.f, 0.f, 0.f, 0.f); q1 = q0; }
        v2h xp[4];
        xp[0] = (v2h){(_Float16)q0.x, (_Float16)q0.y};
        xp[1] = (v2h){(_Float16)q0.z, (_Float16)q0.w};
        xp[2] = (v2h){(_Float16)q1.x, (_Float16)q1.y};
        xp[3] = (v2h){(_Float16)q1.z, (_Float16)q1.w};

        v8f c = {};
#pragma unroll
        for (int kk = 0; kk < NSTEP; ++kk) {
            const int kbase = kk * 32;
            typedef __attribute__((ext_vector_type(8))) _Float16 v8h_t;
            v16h b = __builtin_shufflevector(*(const v8h_t*)(brow + kbase), *(const v8h_t*)(brow + kbase + 16), 0, 1, 2, 3, 4, 5, 6, 7, 8, 9, 10, 11, 12, 13, 14, 15);

            union AU { v16h v; v2h p[8]; } au;
            if (FI == 16) {
                // slots 0-7: k0 = 2kk, i = iofs+0..7 ; slots 8-15: k1 = 2kk+1
                const int k0 = 2 * kk, k1 = 2 * kk + 1;
                _Float16 hh0 = (_Float16)hval(k0, ea0, ea1, w1, b1);
                _Float16 hh1 = (_Float16)hval(k1, ea0, ea1, w1, b1);
                v2h hb0 = (v2h){hh0, hh0};
                v2h hb1 = (v2h){hh1, hh1};
#pragma unroll
                for (int j = 0; j < 4; ++j) {
                    au.p[j]     = hb0 * xp[j];   // v_pk_mul_f16
                    au.p[4 + j] = hb1 * xp[j];
                }
            } else {
                // FI==4: groups of 4 slots share one k; i cycles 0..3
                const int kadd = lhalf ? 2 : 0;
#pragma unroll
                for (int r = 0; r < 2; ++r) {
#pragma unroll
                    for (int g = 0; g < 2; ++g) {
                        const int kc = 8 * kk + 4 * r + g;      // + kadd at runtime
                        int k = kc + kadd;
                        int K0 = kbase + 16 * r + 4 * g + ihofs;
                        int kcl = (k < EHID) ? k : 0;
                        float hraw = fmaxf(fmaf(ea1, w1[EHID + kcl],
                                           fmaf(ea0, w1[kcl], b1[kcl])), 0.0f);
                        float hv = (K0 < KV)
                                 ? ((k < EHID) ? hraw : ((k == EHID) ? 1.0f : 0.0f))
                                 : 0.0f;
                        _Float16 hh = (_Float16)hv;
                        v2h hb = (v2h){hh, hh};
                        au.p[r * 4 + g * 2 + 0] = hb * xp[0];
                        au.p[r * 4 + g * 2 + 1] = hb * xp[1];
                    }
                }
            }
            c = __builtin_amdgcn_wmma_f32_16x16x32_f16(
                    false, au.v, false, b, (short)0, c, false, false);
        }

        // C layout: lane L, vgpr r -> M = r + 8*(L>=16), N = L&15
        if (erow < FO) {
#pragma unroll
            for (int vr = 0; vr < 8; ++vr) {
                int er = t * 16 + (lhalf << 3) + vr;
                if (er < nEdges) {
                    int d = dst[er];
                    atomicAdd(&agg[d * 16 + erow], c[vr]);
                }
            }
        }
    }
}

// ---------------------------------------------------------------------------
// out = agg + x @ root + bias ; optional layernorm+relu ; optional packed out
// ---------------------------------------------------------------------------
template <int FI, int FO, bool LN, bool FINAL>
__global__ void __launch_bounds__(256)
node_update_kernel(const float* __restrict__ xin, const float* __restrict__ agg,
                   const float* __restrict__ root, const float* __restrict__ bias,
                   const float* __restrict__ g, const float* __restrict__ bta,
                   float* __restrict__ xout, int n)
{
    int i = blockIdx.x * blockDim.x + threadIdx.x;
    if (i >= n) return;
    float acc[FO];
#pragma unroll
    for (int o = 0; o < FO; ++o) acc[o] = agg[i * 16 + o] + bias[o];
#pragma unroll
    for (int fi = 0; fi < FI; ++fi) {
        float xv = xin[i * 16 + fi];
#pragma unroll
        for (int o = 0; o < FO; ++o)
            acc[o] = fmaf(xv, root[fi * FO + o], acc[o]);
    }
    if (LN) {
        float m = 0.0f;
#pragma unroll
        for (int o = 0; o < FO; ++o) m += acc[o];
        m *= (1.0f / FO);
        float v = 0.0f;
#pragma unroll
        for (int o = 0; o < FO; ++o) { float d = acc[o] - m; v = fmaf(d, d, v); }
        v *= (1.0f / FO);
        float inv = 1.0f / sqrtf(v + 1e-5f);
#pragma unroll
        for (int o = 0; o < FO; ++o)
            acc[o] = reluf(fmaf((acc[o] - m) * inv, g[o], bta[o]));
    }
    if (FINAL) {
#pragma unroll
        for (int o = 0; o < FO; ++o) xout[i * FO + o] = acc[o];
    } else {
#pragma unroll
        for (int o = 0; o < FO; ++o) xout[i * 16 + o] = acc[o];
#pragma unroll
        for (int o = FO; o < 16; ++o) xout[i * 16 + o] = 0.0f;
    }
}

// ---------------------------------------------------------------------------
// d_in order (setup_inputs insertion order, nested dicts depth-first):
//  0:x 1:pred_mask 2:edge_attr
//  3+6l .. : layer l {w1,b1,w2,b2,root,bias} for l=0..3
//  27+2i  : norms i {g,b} for i=0..2
//  33..36 : mask_embd {w1,b1,w2,b2}
//  37     : edge_index [2, 2*E_DIR] int32
// ---------------------------------------------------------------------------
extern "C" void kernel_launch(void* const* d_in, const int* in_sizes, int n_in,
                              void* d_out, int out_size, void* d_ws, size_t ws_size,
                              hipStream_t stream)
{
    (void)in_sizes; (void)n_in; (void)out_size; (void)ws_size;

    const float* x  = (const float*)d_in[0];
    const float* pm = (const float*)d_in[1];
    const float* ea = (const float*)d_in[2];

    const float *Lw1[4], *Lb1[4], *Lw2[4], *Lb2[4], *Lroot[4], *Lbias[4];
    for (int l = 0; l < 4; ++l) {
        const int b = 3 + 6 * l;
        Lw1[l]   = (const float*)d_in[b + 0];
        Lb1[l]   = (const float*)d_in[b + 1];
        Lw2[l]   = (const float*)d_in[b + 2];
        Lb2[l]   = (const float*)d_in[b + 3];
        Lroot[l] = (const float*)d_in[b + 4];
        Lbias[l] = (const float*)d_in[b + 5];
    }
    const float *Ng[3], *Nb[3];
    for (int i = 0; i < 3; ++i) {
        Ng[i] = (const float*)d_in[27 + 2 * i];
        Nb[i] = (const float*)d_in[28 + 2 * i];
    }
    const float* mw1 = (const float*)d_in[33];
    const float* mb1 = (const float*)d_in[34];
    const float* mw2 = (const float*)d_in[35];
    const float* mb2 = (const float*)d_in[36];
    const int* eidx  = (const int*)d_in[37];
    const int* srcv  = eidx;
    const int* dstv  = eidx + E_EDGES;

    float* ws    = (float*)d_ws;
    float* xcur  = ws;
    float* aggb  = ws + (size_t)N_NODES * 16;
    float* xnext = ws + (size_t)N_NODES * 32;
    float* out   = (float*)d_out;

    const int TB = 256;
    const int nbNodes = (N_NODES + TB - 1) / TB;
    const int nbFill  = (N_NODES * 16 + TB - 1) / TB;
    const int tiles   = (E_EDGES + 15) / 16;
    const int nbEdge  = (tiles + (TB / 32) - 1) / (TB / 32);

    mask_embd_kernel<<<nbNodes, TB, 0, stream>>>(x, pm, mw1, mb1, mw2, mb2,
                                                 xcur, N_NODES);

    // ----- layer 0: 4 -> 16, LN + relu -----
    fill_zero_kernel<<<nbFill, TB, 0, stream>>>(aggb, N_NODES * 16);
    edge_msg_kernel<4, 16><<<nbEdge, TB, 0, stream>>>(
        xcur, srcv, dstv, ea, Lw1[0], Lb1[0], Lw2[0], Lb2[0], aggb, E_EDGES);
    node_update_kernel<4, 16, true, false><<<nbNodes, TB, 0, stream>>>(
        xcur, aggb, Lroot[0], Lbias[0], Ng[0], Nb[0], xnext, N_NODES);
    { float* t = xcur; xcur = xnext; xnext = t; }

    // ----- layers 1,2: 16 -> 16, LN + relu -----
    for (int l = 1; l <= 2; ++l) {
        fill_zero_kernel<<<nbFill, TB, 0, stream>>>(aggb, N_NODES * 16);
        edge_msg_kernel<16, 16><<<nbEdge, TB, 0, stream>>>(
            xcur, srcv, dstv, ea, Lw1[l], Lb1[l], Lw2[l], Lb2[l], aggb, E_EDGES);
        node_update_kernel<16, 16, true, false><<<nbNodes, TB, 0, stream>>>(
            xcur, aggb, Lroot[l], Lbias[l], Ng[l], Nb[l], xnext, N_NODES);
        { float* t = xcur; xcur = xnext; xnext = t; }
    }

    // ----- layer 3: 16 -> 4, no LN, packed output -----
    fill_zero_kernel<<<nbFill, TB, 0, stream>>>(aggb, N_NODES * 16);
    edge_msg_kernel<16, 4><<<nbEdge, TB, 0, stream>>>(
        xcur, srcv, dstv, ea, Lw1[3], Lb1[3], Lw2[3], Lb2[3], aggb, E_EDGES);
    node_update_kernel<16, 4, false, true><<<nbNodes, TB, 0, stream>>>(
        xcur, aggb, Lroot[3], Lbias[3], nullptr, nullptr, out, N_NODES);
}
